// RepPointsHead_90417651516405
// MI455X (gfx1250) — matrix operand order repaired and checked by reading, not compiled, not run on hardware
//
#include <hip/hip_runtime.h>

// ---------------------------------------------------------------------------
// RepPoints head forward for MI455X (gfx1250), bf16 WMMA implicit-GEMM convs.
// Round 4: async-LDS builtin wants (v4i* src, as3 v4i* dst, imm off, imm cpol).
// ---------------------------------------------------------------------------

#define C_FEAT 256
#define HH 100
#define WW 152
#define HWSZ (HH * WW)          // 15200
#define NBATCH 2
#define NS_TOT (NBATCH * HWSZ)  // 30400 spatial positions
#define KTOT (C_FEAT * 9)       // 2304 GEMM K
#define NS_BLK 238              // ceil(30400 / 128)

#if __has_builtin(__builtin_amdgcn_global_load_async_to_lds_b128)
#define HAVE_ASYNC_LDS 1
#else
#define HAVE_ASYNC_LDS 0
#endif

#if HAVE_ASYNC_LDS
#if __has_builtin(__builtin_amdgcn_s_wait_asynccnt)
#define WAIT_ASYNC() __builtin_amdgcn_s_wait_asynccnt(0)
#else
#define WAIT_ASYNC() asm volatile("s_wait_asynccnt 0x0" ::: "memory")
#endif
#else
#define WAIT_ASYNC() ((void)0)
#endif

typedef __attribute__((ext_vector_type(16))) __bf16 bf16x16;
typedef __attribute__((ext_vector_type(8)))  float  f32x8;
// exact parameter types reported by clang for the async-LDS builtin
typedef __attribute__((__vector_size__(4 * sizeof(int)))) int v4i_async;
typedef __attribute__((address_space(3))) v4i_async* lds_v4i_ptr;
typedef __attribute__((address_space(3))) void*      lds_void_ptr;

union Frag { bf16x16 v; uint4 q[2]; };

__device__ inline unsigned short f2bf(float f) {
  unsigned int u = __float_as_uint(f);
  unsigned int r = u + 0x7FFFu + ((u >> 16) & 1u);   // round-to-nearest-even
  return (unsigned short)(r >> 16);
}
__device__ inline float bf2f(unsigned short h) {
  return __uint_as_float(((unsigned int)h) << 16);
}

// 32-byte tile-row copy: global -> LDS (async on CDNA5 when available)
__device__ inline void stage32B(const unsigned short* __restrict__ src,
                                unsigned short* __restrict__ dst) {
#if HAVE_ASYNC_LDS
  lds_v4i_ptr ldst = (lds_v4i_ptr)(lds_void_ptr)dst;
  __builtin_amdgcn_global_load_async_to_lds_b128((v4i_async*)src, ldst, 0, 0);
  // IOFFSET is added to BOTH the global and LDS addresses.
  __builtin_amdgcn_global_load_async_to_lds_b128((v4i_async*)src, ldst, 16, 0);
#else
  ((uint4*)dst)[0] = ((const uint4*)src)[0];
  ((uint4*)dst)[1] = ((const uint4*)src)[1];
#endif
}

// -------------------- elementwise converters --------------------------------
__global__ void cvt_f32_bf16_k(const float* __restrict__ in,
                               unsigned short* __restrict__ out, int n) {
  int i = blockIdx.x * 256 + threadIdx.x;
  if (i < n) out[i] = f2bf(in[i]);
}

__global__ void relu_f32_bf16_k(const float* __restrict__ in,
                                unsigned short* __restrict__ out, int n) {
  int i = blockIdx.x * 256 + threadIdx.x;
  if (i < n) { float v = in[i]; out[i] = f2bf(v > 0.f ? v : 0.f); }
}

// -------------------- 3x3 conv as implicit GEMM (bf16 WMMA) ----------------
// M = out channels (256), N = spatial (B*H*W), K = Cin*9.
// blockIdx.x -> spatial tile (128), blockIdx.y -> channel tile (128).
__global__ __launch_bounds__(256)
void conv3x3_wmma_k(const unsigned short* __restrict__ xbf,  // [B][C][H][W] bf16
                    const unsigned short* __restrict__ wp,   // [256][2304]  bf16
                    const float* __restrict__ bias,          // [256] or null
                    float* __restrict__ out) {               // [B][256][H][W]
  __shared__ __align__(16) unsigned short tA[2][128 * 32];
  __shared__ __align__(16) unsigned short tB[2][128 * 32];

  const int tid  = threadIdx.x;
  const int lane = tid & 31;
  const int wv   = tid >> 5;
  const int wm   = wv & 3;          // 4 waves across M
  const int wn   = wv >> 2;         // 2 waves across N
  const int Nb   = blockIdx.x * 128;
  const int Mb   = blockIdx.y * 128;
  const int hi   = lane >> 4, lr = lane & 15;

  f32x8 acc[2][4];
  for (int i = 0; i < 2; ++i)
    for (int j = 0; j < 4; ++j)
      for (int r = 0; r < 8; ++r) acc[i][j][r] = 0.f;

  const int fm   = tid >> 1;        // 0..127 (tile row for fills)
  const int half = tid & 1;         // 0/1    (16-element half of K slice)

  // decode this thread's spatial position once
  int s = Nb + fm;
  int sb = 0, sh = 0, sw = 0;
  bool sok = s < NS_TOT;
  if (sok) { sb = s / HWSZ; int rem = s % HWSZ; sh = rem / WW; sw = rem % WW; }
  const unsigned short* xplane = xbf + (size_t)sb * C_FEAT * HWSZ;
  const unsigned short* wrow   = wp + (size_t)(Mb + fm) * KTOT;

  auto fillA = [&](int buf, int kc) {
    stage32B(wrow + kc + half * 16, &tA[buf][fm * 32 + half * 16]);
    if (kc + 32 < KTOT) __builtin_prefetch(wrow + kc + 32, 0, 3);
  };
  auto fillB = [&](int buf, int kc) {
    __align__(16) unsigned short vals[16];
    if (sok) {
      for (int j = 0; j < 16; ++j) {
        int k  = kc + half * 16 + j;
        int ci = k / 9, kp = k % 9;
        int hy = sh + kp / 3 - 1;
        int wx = sw + kp % 3 - 1;
        unsigned short v = 0;
        if ((unsigned)hy < (unsigned)HH && (unsigned)wx < (unsigned)WW)
          v = xplane[(size_t)ci * HWSZ + hy * WW + wx];
        vals[j] = v;
      }
    } else {
      for (int j = 0; j < 16; ++j) vals[j] = 0;
    }
    uint4* dst = (uint4*)&tB[buf][fm * 32 + half * 16];
    dst[0] = *(const uint4*)&vals[0];
    dst[1] = *(const uint4*)&vals[8];
  };

  // prologue: stage K-slice 0 into buffer 0
  fillA(0, 0);
  fillB(0, 0);
  WAIT_ASYNC();
  __syncthreads();

  int cur = 0;
  for (int kc = 0; kc < KTOT; kc += 32) {
    const int nxt = cur ^ 1;
    if (kc + 32 < KTOT) {   // stage next slice while computing current
      fillA(nxt, kc + 32);
      fillB(nxt, kc + 32);
    }
    // ---- WMMA over current K slice --------------------------------------
    {
      Frag a[2];
      for (int i = 0; i < 2; ++i) {
        const uint4* rowA = (const uint4*)&tA[cur][(wm * 32 + i * 16 + lr) * 32];
        a[i].q[0] = rowA[hi * 2];      // k = 16*hi + 0..7
        a[i].q[1] = rowA[hi * 2 + 1];  // k = 16*hi + 8..15
      }
      for (int j = 0; j < 4; ++j) {
        Frag bq;
        const uint4* rowB = (const uint4*)&tB[cur][(wn * 64 + j * 16 + lr) * 32];
        bq.q[0] = rowB[hi * 2];        // k = 16*hi + 0..7
        bq.q[1] = rowB[hi * 2 + 1];    // k = 16*hi + 8..15
        for (int i = 0; i < 2; ++i)
          acc[i][j] = __builtin_amdgcn_wmma_f32_16x16x32_bf16(
              false, a[i].v, false, bq.v, (short)0, acc[i][j], false, false);
      }
    }
    WAIT_ASYNC();          // publish async LDS writes before the barrier
    __syncthreads();
    cur = nxt;
  }

  // ---- epilogue: bias + store -------------------------------------------
  for (int j = 0; j < 4; ++j) {
    int sp = Nb + wn * 64 + j * 16 + lr;
    if (sp >= NS_TOT) continue;
    int b = sp / HWSZ, hw = sp % HWSZ;
    for (int i = 0; i < 2; ++i) {
      for (int r = 0; r < 8; ++r) {
        int co = Mb + wm * 32 + i * 16 + r + 8 * hi;
        float v = acc[i][j][r];
        if (bias) v += bias[co];
        out[((size_t)b * C_FEAT + co) * HWSZ + hw] = v;
      }
    }
  }
}

// -------------------- GroupNorm (32 groups) + ReLU -> bf16 ------------------
__global__ __launch_bounds__(256)
void gn_relu_bf16_k(const float* __restrict__ in,
                    const float* __restrict__ gamma,
                    const float* __restrict__ beta,
                    unsigned short* __restrict__ outbf) {
  const int bg = blockIdx.x;         // [0, B*32)
  const int b = bg / 32, g = bg % 32;
  const int CH = C_FEAT / 32;        // 8
  const int NEL = CH * HWSZ;         // 121600
  const float* base = in + ((size_t)b * C_FEAT + g * CH) * HWSZ;

  float s = 0.f, ss = 0.f;
  for (int i = threadIdx.x; i < NEL; i += 256) {
    float v = base[i];
    s += v; ss += v * v;
  }
  __shared__ float rs[256], rss[256];
  rs[threadIdx.x] = s; rss[threadIdx.x] = ss;
  __syncthreads();
  for (int off = 128; off > 0; off >>= 1) {
    if (threadIdx.x < off) {
      rs[threadIdx.x]  += rs[threadIdx.x + off];
      rss[threadIdx.x] += rss[threadIdx.x + off];
    }
    __syncthreads();
  }
  const float mu  = rs[0] / (float)NEL;
  const float var = rss[0] / (float)NEL - mu * mu;
  const float inv = rsqrtf(var + 1e-5f);

  unsigned short* ob = outbf + ((size_t)b * C_FEAT + g * CH) * HWSZ;
  for (int i = threadIdx.x; i < NEL; i += 256) {
    int c = g * CH + i / HWSZ;
    float v = (base[i] - mu) * inv * gamma[c] + beta[c];
    ob[i] = f2bf(v > 0.f ? v : 0.f);
  }
}

// -------------------- 1x1 conv (small N heads) ------------------------------
__global__ void conv1x1_k(const unsigned short* __restrict__ inbf, // [B][256][HW]
                          const float* __restrict__ w,             // [O][256]
                          const float* __restrict__ bias,          // [O]
                          int O,
                          const float* __restrict__ addend,        // null or [B][O][HW]
                          float* __restrict__ out) {
  int idx = blockIdx.x * 256 + threadIdx.x;
  int total = NBATCH * O * HWSZ;
  if (idx >= total) return;
  int b = idx / (O * HWSZ);
  int rem = idx % (O * HWSZ);
  int o = rem / HWSZ, hw = rem % HWSZ;
  const unsigned short* xb = inbf + (size_t)b * C_FEAT * HWSZ + hw;
  const float* wr = w + (size_t)o * C_FEAT;
  float acc = bias ? bias[o] : 0.f;
  for (int c = 0; c < C_FEAT; ++c)
    acc += bf2f(xb[(size_t)c * HWSZ]) * wr[c];
  if (addend) acc += addend[idx];
  out[idx] = acc;
}

// -------------------- dcn offset = pts_init - base --------------------------
__global__ void make_offset_k(const float* __restrict__ pinit,
                              float* __restrict__ off) {
  int idx = blockIdx.x * 256 + threadIdx.x;
  int total = NBATCH * 18 * HWSZ;
  if (idx >= total) return;
  int ch = (idx / HWSZ) % 18;
  int kp = ch >> 1;
  float basev = (ch & 1) ? (float)(kp % 3 - 1) : (float)(kp / 3 - 1);
  off[idx] = pinit[idx] - basev;
}

// -------------------- deformable conv as implicit GEMM ----------------------
struct DcnEnt { float w00, w01, w10, w11; int i00, i01, i10, i11; int plane; };

__global__ __launch_bounds__(256)
void dcn_wmma_k(const unsigned short* __restrict__ feat, // [B][256][H][W] bf16
                const float* __restrict__ off,           // [B][18][H][W]
                const unsigned short* __restrict__ wp,   // [256][2304] bf16
                float* __restrict__ out) {               // [B][256][H][W]
  __shared__ __align__(16) unsigned short tA[2][128 * 32];
  __shared__ __align__(16) unsigned short tB[2][128 * 32];
  __shared__ DcnEnt tab[128 * 9];

  const int tid  = threadIdx.x;
  const int lane = tid & 31;
  const int wv   = tid >> 5;
  const int wm   = wv & 3;
  const int wn   = wv >> 2;
  const int Nb   = blockIdx.x * 128;
  const int Mb   = blockIdx.y * 128;
  const int hi   = lane >> 4, lr = lane & 15;

  // ---- precompute bilinear corners/weights per (position, kernel point) ---
  for (int t = tid; t < 128 * 9; t += 256) {
    int n = t / 9, kp = t % 9;
    int s = Nb + n;
    DcnEnt e;
    e.w00 = e.w01 = e.w10 = e.w11 = 0.f;
    e.i00 = e.i01 = e.i10 = e.i11 = 0;
    e.plane = 0;
    if (s < NS_TOT) {
      int b = s / HWSZ, rem = s % HWSZ, h = rem / WW, w = rem % WW;
      float oy = off[((size_t)b * 18 + kp * 2)     * HWSZ + rem];
      float ox = off[((size_t)b * 18 + kp * 2 + 1) * HWSZ + rem];
      float gy = (float)h + (float)(kp / 3 - 1) + oy;
      float gx = (float)w + (float)(kp % 3 - 1) + ox;
      float y0f = floorf(gy), x0f = floorf(gx);
      float dy = gy - y0f, dx = gx - x0f;
      int y0 = (int)y0f, x0 = (int)x0f;
      int y1 = y0 + 1, x1 = x0 + 1;
      bool vy0 = (unsigned)y0 < (unsigned)HH, vy1 = (unsigned)y1 < (unsigned)HH;
      bool vx0 = (unsigned)x0 < (unsigned)WW, vx1 = (unsigned)x1 < (unsigned)WW;
      int y0c = y0 < 0 ? 0 : (y0 > HH - 1 ? HH - 1 : y0);
      int y1c = y1 < 0 ? 0 : (y1 > HH - 1 ? HH - 1 : y1);
      int x0c = x0 < 0 ? 0 : (x0 > WW - 1 ? WW - 1 : x0);
      int x1c = x1 < 0 ? 0 : (x1 > WW - 1 ? WW - 1 : x1);
      e.w00 = (vy0 && vx0) ? (1.f - dy) * (1.f - dx) : 0.f;
      e.w01 = (vy0 && vx1) ? (1.f - dy) * dx : 0.f;
      e.w10 = (vy1 && vx0) ? dy * (1.f - dx) : 0.f;
      e.w11 = (vy1 && vx1) ? dy * dx : 0.f;
      e.i00 = y0c * WW + x0c; e.i01 = y0c * WW + x1c;
      e.i10 = y1c * WW + x0c; e.i11 = y1c * WW + x1c;
      e.plane = b * C_FEAT * HWSZ;
    }
    tab[t] = e;
  }

  f32x8 acc[2][4];
  for (int i = 0; i < 2; ++i)
    for (int j = 0; j < 4; ++j)
      for (int r = 0; r < 8; ++r) acc[i][j][r] = 0.f;

  const int fm = tid >> 1, half = tid & 1;
  const unsigned short* wrow = wp + (size_t)(Mb + fm) * KTOT;

  auto fillA = [&](int buf, int kc) {
    stage32B(wrow + kc + half * 16, &tA[buf][fm * 32 + half * 16]);
    if (kc + 32 < KTOT) __builtin_prefetch(wrow + kc + 32, 0, 3);
  };
  auto fillB = [&](int buf, int kc) {
    __align__(16) unsigned short vals[16];
    for (int j = 0; j < 16; ++j) {
      int k  = kc + half * 16 + j;
      int ci = k / 9, kp = k % 9;
      const DcnEnt& e = tab[fm * 9 + kp];
      const unsigned short* p = feat + (size_t)e.plane + (size_t)ci * HWSZ;
      float v = e.w00 * bf2f(p[e.i00]) + e.w01 * bf2f(p[e.i01]) +
                e.w10 * bf2f(p[e.i10]) + e.w11 * bf2f(p[e.i11]);
      vals[j] = f2bf(v);
    }
    uint4* dst = (uint4*)&tB[buf][fm * 32 + half * 16];
    dst[0] = *(const uint4*)&vals[0];
    dst[1] = *(const uint4*)&vals[8];
  };

  // barrier covers both the tab[] table and the first staged slice
  fillA(0, 0);
  __syncthreads();          // tab ready (fillB reads it)
  fillB(0, 0);
  WAIT_ASYNC();
  __syncthreads();

  int cur = 0;
  for (int kc = 0; kc < KTOT; kc += 32) {
    const int nxt = cur ^ 1;
    if (kc + 32 < KTOT) {
      fillA(nxt, kc + 32);
      fillB(nxt, kc + 32);
    }
    {
      Frag a[2];
      for (int i = 0; i < 2; ++i) {
        const uint4* rowA = (const uint4*)&tA[cur][(wm * 32 + i * 16 + lr) * 32];
        a[i].q[0] = rowA[hi * 2];
        a[i].q[1] = rowA[hi * 2 + 1];
      }
      for (int j = 0; j < 4; ++j) {
        Frag bq;
        const uint4* rowB = (const uint4*)&tB[cur][(wn * 64 + j * 16 + lr) * 32];
        bq.q[0] = rowB[hi * 2];
        bq.q[1] = rowB[hi * 2 + 1];
        for (int i = 0; i < 2; ++i)
          acc[i][j] = __builtin_amdgcn_wmma_f32_16x16x32_bf16(
              false, a[i].v, false, bq.v, (short)0, acc[i][j], false, false);
      }
    }
    WAIT_ASYNC();
    __syncthreads();
    cur = nxt;
  }

  for (int j = 0; j < 4; ++j) {
    int sp = Nb + wn * 64 + j * 16 + lr;
    if (sp >= NS_TOT) continue;
    int b = sp / HWSZ, hw = sp % HWSZ;
    for (int i = 0; i < 2; ++i) {
      for (int r = 0; r < 8; ++r) {
        int co = Mb + wm * 32 + i * 16 + r + 8 * hi;
        out[((size_t)b * C_FEAT + co) * HWSZ + hw] = acc[i][j][r];
      }
    }
  }
}

// ---------------------------------------------------------------------------
extern "C" void kernel_launch(void* const* d_in, const int* in_sizes, int n_in,
                              void* d_out, int out_size, void* d_ws, size_t ws_size,
                              hipStream_t stream) {
  const float* x           = (const float*)d_in[0];
  const float* cls_w       = (const float*)d_in[1];
  const float* cls_b       = (const float*)d_in[2];
  const float* cls_gn_g    = (const float*)d_in[3];
  const float* cls_gn_b    = (const float*)d_in[4];
  const float* reg_w       = (const float*)d_in[5];
  const float* reg_b       = (const float*)d_in[6];
  const float* reg_gn_g    = (const float*)d_in[7];
  const float* reg_gn_b    = (const float*)d_in[8];
  const float* init_conv_w = (const float*)d_in[9];
  const float* init_conv_b = (const float*)d_in[10];
  const float* init_out_w  = (const float*)d_in[11];
  const float* init_out_b  = (const float*)d_in[12];
  const float* cls_dcn_w   = (const float*)d_in[13];
  const float* cls_out_w   = (const float*)d_in[14];
  const float* cls_out_b   = (const float*)d_in[15];
  const float* ref_dcn_w   = (const float*)d_in[16];
  const float* ref_out_w   = (const float*)d_in[17];
  const float* ref_out_b   = (const float*)d_in[18];

  const size_t NA  = (size_t)NBATCH * C_FEAT * HWSZ;   // 7,782,400
  const size_t WEL = (size_t)C_FEAT * KTOT;            // 589,824 per 3x3 conv

  char* ws = (char*)d_ws;
  size_t off = 0;
  auto carve = [&](size_t bytes) -> void* {
    off = (off + 255) & ~(size_t)255;
    void* p = ws + off;
    off += bytes;
    return p;
  };

  unsigned short* xbf    = (unsigned short*)carve(NA * 2);
  unsigned short* A1     = (unsigned short*)carve(NA * 2);
  unsigned short* A2     = (unsigned short*)carve(NA * 2);
  unsigned short* A3     = (unsigned short*)carve(NA * 2);
  unsigned short* A4     = (unsigned short*)carve(NA * 2);
  float*          cf32   = (float*)carve(NA * 4);
  unsigned short* clswp  = (unsigned short*)carve(3 * WEL * 2);
  unsigned short* regwp  = (unsigned short*)carve(3 * WEL * 2);
  unsigned short* initwp = (unsigned short*)carve(WEL * 2);
  unsigned short* cdcnwp = (unsigned short*)carve(WEL * 2);
  unsigned short* rdcnwp = (unsigned short*)carve(WEL * 2);
  float*          dcnoff = (float*)carve((size_t)NBATCH * 18 * HWSZ * 4);
  if (off > ws_size) return;  // workspace too small; cannot proceed safely

  float* out_cls  = (float*)d_out;                              // [2,80,100,152]
  float* out_init = out_cls + (size_t)NBATCH * 80 * HWSZ;       // [2,18,100,152]
  float* out_ref  = out_init + (size_t)NBATCH * 18 * HWSZ;      // [2,18,100,152]

  auto blk = [](size_t n) { return (unsigned)((n + 255) / 256); };

  // ---- fp32 -> bf16 conversions -----------------------------------------
  cvt_f32_bf16_k<<<blk(NA), 256, 0, stream>>>(x, xbf, (int)NA);
  cvt_f32_bf16_k<<<blk(3 * WEL), 256, 0, stream>>>(cls_w, clswp, (int)(3 * WEL));
  cvt_f32_bf16_k<<<blk(3 * WEL), 256, 0, stream>>>(reg_w, regwp, (int)(3 * WEL));
  cvt_f32_bf16_k<<<blk(WEL), 256, 0, stream>>>(init_conv_w, initwp, (int)WEL);
  cvt_f32_bf16_k<<<blk(WEL), 256, 0, stream>>>(cls_dcn_w, cdcnwp, (int)WEL);
  cvt_f32_bf16_k<<<blk(WEL), 256, 0, stream>>>(ref_dcn_w, rdcnwp, (int)WEL);

  dim3 gconv(NS_BLK, 2);

  // ---- conv towers: conv3x3 -> GN -> ReLU, three stages each -------------
  unsigned short* cbuf[2] = {A1, A2};
  unsigned short* pbuf[2] = {A3, A4};
  const unsigned short* ccur = xbf;
  const unsigned short* pcur = xbf;
  for (int i = 0; i < 3; ++i) {
    unsigned short* cdst = cbuf[i & 1];
    conv3x3_wmma_k<<<gconv, 256, 0, stream>>>(ccur, clswp + (size_t)i * WEL,
                                              cls_b + i * C_FEAT, cf32);
    gn_relu_bf16_k<<<NBATCH * 32, 256, 0, stream>>>(cf32, cls_gn_g + i * C_FEAT,
                                                    cls_gn_b + i * C_FEAT, cdst);
    ccur = cdst;

    unsigned short* pdst = pbuf[i & 1];
    conv3x3_wmma_k<<<gconv, 256, 0, stream>>>(pcur, regwp + (size_t)i * WEL,
                                              reg_b + i * C_FEAT, cf32);
    gn_relu_bf16_k<<<NBATCH * 32, 256, 0, stream>>>(cf32, reg_gn_g + i * C_FEAT,
                                                    reg_gn_b + i * C_FEAT, pdst);
    pcur = pdst;
  }
  // final: ccur == A1 (cls feat), pcur == A3 (pts feat)

  // ---- pts_out_init = 1x1(relu(conv3x3(pts_feat))) -----------------------
  conv3x3_wmma_k<<<gconv, 256, 0, stream>>>(pcur, initwp, init_conv_b, cf32);
  relu_f32_bf16_k<<<blk(NA), 256, 0, stream>>>(cf32, A4, (int)NA);
  conv1x1_k<<<blk((size_t)NBATCH * 18 * HWSZ), 256, 0, stream>>>(
      A4, init_out_w, init_out_b, 18, nullptr, out_init);

  // ---- dcn_offset = pts_out_init - base (forward of gradient_mul trick) --
  make_offset_k<<<blk((size_t)NBATCH * 18 * HWSZ), 256, 0, stream>>>(out_init, dcnoff);

  // ---- cls branch: deform conv -> relu -> 1x1 ----------------------------
  dcn_wmma_k<<<gconv, 256, 0, stream>>>(ccur, dcnoff, cdcnwp, cf32);
  relu_f32_bf16_k<<<blk(NA), 256, 0, stream>>>(cf32, A2, (int)NA);
  conv1x1_k<<<blk((size_t)NBATCH * 80 * HWSZ), 256, 0, stream>>>(
      A2, cls_out_w, cls_out_b, 80, nullptr, out_cls);

  // ---- refine branch: deform conv -> relu -> 1x1 (+ pts_out_init) --------
  dcn_wmma_k<<<gconv, 256, 0, stream>>>(pcur, dcnoff, rdcnwp, cf32);
  relu_f32_bf16_k<<<blk(NA), 256, 0, stream>>>(cf32, A4, (int)NA);
  conv1x1_k<<<blk((size_t)NBATCH * 18 * HWSZ), 256, 0, stream>>>(
      A4, ref_out_w, ref_out_b, 18, out_init, out_ref);
}
